// SingleStreamAttention_39273180954864
// MI455X (gfx1250) — matrix operand order repaired and checked by reading, not compiled, not run on hardware
//
#include <hip/hip_runtime.h>
#include <hip/hip_bf16.h>

// ---------------------------------------------------------------- types
typedef __attribute__((ext_vector_type(16))) __bf16       v16bf;
typedef __attribute__((ext_vector_type(8)))  float        v8f;
typedef __attribute__((ext_vector_type(4)))  unsigned int v4u;
typedef __attribute__((ext_vector_type(4)))  float        v4f;
typedef __attribute__((ext_vector_type(4)))  __bf16       v4bf;

union Frag { v16bf v; v4u q[2]; };

// ---------------------------------------------------------------- shapes
#define DIM_C   2048
#define HEADS_N 16
#define HD_N    128
#define S_LEN   1560
#define BN_N    21
#define NTOK    (BN_N * S_LEN)   // 32760
#define NA_N    32
#define ENC_N   768
#define KVN_N   4096

// GEMM tiling: block 128x256, 8 waves in 2(M) x 4(N), wave tile 64x64
#define BM   128
#define BNT  256
#define KB   32
#define APAD 40   // padded LDS stride (bf16 elems): 80B rows -> 16B-aligned frag loads

__device__ __forceinline__ __bf16 f2bf(float f) {
  unsigned u = __builtin_bit_cast(unsigned, f);
  unsigned r = (u + 0x7FFFu + ((u >> 16) & 1u)) >> 16;
  unsigned short s = (unsigned short)r;
  return __builtin_bit_cast(__bf16, s);
}

__device__ __forceinline__ v8f wmma_bf16(const Frag& a, const Frag& b, v8f c) {
  return __builtin_amdgcn_wmma_f32_16x16x32_bf16(false, a.v, false, b.v,
                                                 (short)0, c, false, false);
}

// async memory->LDS 16B copy per lane (CDNA5 async engine, ASYNCcnt)
__device__ __forceinline__ void async_copy_b128(const __bf16* gsrc, const __bf16* ldst) {
  unsigned lds = (unsigned)(uintptr_t)ldst;
  unsigned long long ga = (unsigned long long)(uintptr_t)gsrc;
  asm volatile("global_load_async_to_lds_b128 %0, %1, off"
               :: "v"(lds), "v"(ga) : "memory");
}
__device__ __forceinline__ void wait_async_0() {
  asm volatile("s_wait_asynccnt 0x0" ::: "memory");
}
__device__ __forceinline__ void wait_async_6() {
  asm volatile("s_wait_asynccnt 0x6" ::: "memory");
}

// ---------------------------------------------------------------- fp32 -> bf16 (row-major copy)
__global__ __launch_bounds__(256)
void cvt_f32_bf16_kernel(const float* __restrict__ src, __bf16* __restrict__ dst,
                         long long n) {
  long long i = ((long long)blockIdx.x * 256 + threadIdx.x) * 4;
  if (i + 4 <= n) {
    v4f x = *(const v4f*)&src[i];
    v4bf o;
#pragma unroll
    for (int j = 0; j < 4; ++j) o[j] = f2bf(x[j]);
    *(v4bf*)&dst[i] = o;
  } else {
    for (long long j = i; j < n; ++j) dst[j] = f2bf(src[j]);
  }
}

// ---------------------------------------------------------------- fp32 [K][N] -> bf16 [N][K] (weights)
__global__ __launch_bounds__(256)
void transpose_cvt_kernel(const float* __restrict__ src, __bf16* __restrict__ dst,
                          int K, int N) {
  __shared__ float t[32][33];
  int n0 = blockIdx.x * 32, k0 = blockIdx.y * 32;
  int x = threadIdx.x & 31, y = threadIdx.x >> 5;   // y in 0..7
#pragma unroll
  for (int j = 0; j < 4; ++j) {
    int row = y + j * 8;
    t[row][x] = src[(size_t)(k0 + row) * N + n0 + x];
  }
  __syncthreads();
#pragma unroll
  for (int j = 0; j < 4; ++j) {
    int row = y + j * 8;
    dst[(size_t)(n0 + row) * K + k0 + x] = f2bf(t[x][row]);
  }
}

// ---------------------------------------------------------------- min/max of attn map rows
__global__ __launch_bounds__(1024)
void minmax_kernel(const float* __restrict__ m, int n, float* __restrict__ out4) {
  __shared__ float sm[4][32];
  int tid = threadIdx.x, lane = tid & 31, wave = tid >> 5;
  float mn0 = 3.4e38f, mx0 = -3.4e38f, mn1 = 3.4e38f, mx1 = -3.4e38f;
  for (int i = tid; i < n; i += 1024) {
    float a = m[i], b = m[n + i];
    mn0 = fminf(mn0, a); mx0 = fmaxf(mx0, a);
    mn1 = fminf(mn1, b); mx1 = fmaxf(mx1, b);
  }
#pragma unroll
  for (int o = 16; o > 0; o >>= 1) {
    mn0 = fminf(mn0, __shfl_xor(mn0, o, 32));
    mx0 = fmaxf(mx0, __shfl_xor(mx0, o, 32));
    mn1 = fminf(mn1, __shfl_xor(mn1, o, 32));
    mx1 = fmaxf(mx1, __shfl_xor(mx1, o, 32));
  }
  if (lane == 0) { sm[0][wave] = mn0; sm[1][wave] = mx0; sm[2][wave] = mn1; sm[3][wave] = mx1; }
  __syncthreads();
  if (wave == 0) {
    mn0 = sm[0][lane]; mx0 = sm[1][lane]; mn1 = sm[2][lane]; mx1 = sm[3][lane];
#pragma unroll
    for (int o = 16; o > 0; o >>= 1) {
      mn0 = fminf(mn0, __shfl_xor(mn0, o, 32));
      mx0 = fmaxf(mx0, __shfl_xor(mx0, o, 32));
      mn1 = fminf(mn1, __shfl_xor(mn1, o, 32));
      mx1 = fmaxf(mx1, __shfl_xor(mx1, o, 32));
    }
    if (lane == 0) { out4[0] = mn0; out4[1] = mx0; out4[2] = mn1; out4[3] = mx1; }
  }
}

__global__ __launch_bounds__(256)
void pos_kernel(const float* __restrict__ m, const float* __restrict__ mm,
                float* __restrict__ pos, int n) {
  int i = blockIdx.x * 256 + threadIdx.x;
  if (i >= n) return;
  float a = m[i], b = m[n + i];
  float h1 = (a - mm[0]) / (mm[1] - mm[0]) * 4.0f;          // [0,4]
  float h2 = (b - mm[2]) / (mm[3] - mm[2]) * 4.0f + 20.0f;  // [20,24]
  pos[i] = (a >= b) ? h1 : h2;                              // argmax==0 -> human1
}

// ---------------------------------------------------------------- bf16 WMMA GEMM, fp32 out + bias
// C[M,N] = A[M,K] @ Bt[N,K]^T + bias[N]   (B pre-transposed; double-buffered async LDS fills)
// block 128x256, wave tile 64x64 (4x4 WMMA accumulators), 16 wmma : 16 ds_b128 per K-step
__global__ __launch_bounds__(256)
void gemm_bf16_wmma_kernel(const __bf16* __restrict__ A, const __bf16* __restrict__ Bt,
                           const float* __restrict__ bias, float* __restrict__ C,
                           int M, int N, int K) {
  __shared__ __align__(16) __bf16 As[2][BM * APAD];    // [row][k]
  __shared__ __align__(16) __bf16 Bs[2][BNT * APAD];   // [col][k]
  const int tid  = threadIdx.x;
  const int lane = tid & 31, wave = tid >> 5;
  const int wm = wave & 1, wn = wave >> 1;             // 2 waves in M x 4 waves in N
  const int r = lane & 15, kg = lane >> 4;
  const int m0 = blockIdx.x * BM;
  const int n0 = blockIdx.y * BNT;

  const v8f zacc = {0.f, 0.f, 0.f, 0.f, 0.f, 0.f, 0.f, 0.f};
  v8f acc[4][4];
#pragma unroll
  for (int i = 0; i < 4; ++i)
#pragma unroll
    for (int j = 0; j < 4; ++j) acc[i][j] = zacc;

  // 6 async ops per thread per tile (2 A chunks + 4 B chunks of 8 bf16)
  auto issue_tile = [&](int kb, int buf) {
#pragma unroll
    for (int t = 0; t < 2; ++t) {
      int s = tid + t * 256;            // 512 chunks: 128 rows x 4
      int row = s >> 2, ko = (s & 3) * 8;
      int gm = m0 + row; gm = gm < M ? gm : M - 1;
      async_copy_b128(&A[(size_t)gm * K + kb + ko], &As[buf][row * APAD + ko]);
    }
#pragma unroll
    for (int t = 0; t < 4; ++t) {
      int s = tid + t * 256;            // 1024 chunks: 256 cols x 4
      int col = s >> 2, ko = (s & 3) * 8;
      async_copy_b128(&Bt[(size_t)(n0 + col) * K + kb + ko], &Bs[buf][col * APAD + ko]);
    }
  };

  issue_tile(0, 0);
  const int nk = K / KB;
  for (int i = 0; i < nk; ++i) {
    const int buf = i & 1;
    if (i + 1 < nk) {
      issue_tile((i + 1) * KB, buf ^ 1);
      wait_async_6();                   // own 6 ops of tile i done; next 6 in flight
    } else {
      wait_async_0();
    }
    __syncthreads();                    // all waves' tile-i LDS data visible

    Frag a[4], b[4];
#pragma unroll
    for (int mt = 0; mt < 4; ++mt) {
      int row = wm * 64 + mt * 16 + r;
      a[mt].q[0] = *(const v4u*)&As[buf][row * APAD + 8 * kg];       // e0..7 : K=8*kg+e
      a[mt].q[1] = *(const v4u*)&As[buf][row * APAD + 16 + 8 * kg];  // e8..15: K=16+8*kg+e
    }
#pragma unroll
    for (int nt = 0; nt < 4; ++nt) {
      int col = wn * 64 + nt * 16 + r;
      b[nt].q[0] = *(const v4u*)&Bs[buf][col * APAD + 8 * kg];       // e0..7 : K=8*kg+e
      b[nt].q[1] = *(const v4u*)&Bs[buf][col * APAD + 16 + 8 * kg];  // e8..15: K=16+8*kg+e
    }
#pragma unroll
    for (int mt = 0; mt < 4; ++mt)
#pragma unroll
      for (int nt = 0; nt < 4; ++nt)
        acc[mt][nt] = wmma_bf16(a[mt], b[nt], acc[mt][nt]);

    __syncthreads();                    // done reading buf before it is refilled
  }

  // epilogue: C layout -> lane holds col r, rows v + 8*kg
#pragma unroll
  for (int mt = 0; mt < 4; ++mt) {
#pragma unroll
    for (int nt = 0; nt < 4; ++nt) {
      int gn = n0 + wn * 64 + nt * 16 + r;
      float bv = bias[gn];
#pragma unroll
      for (int v = 0; v < 8; ++v) {
        int gm = m0 + wm * 64 + mt * 16 + v + 8 * kg;
        if (gm < M) C[(size_t)gm * N + gn] = acc[mt][nt][v] + bv;
      }
    }
  }
}

// ---------------------------------------------------------------- Q: RMSNorm + RoPE -> bf16 [bn,h,s,d]
__global__ __launch_bounds__(256)
void qnorm_rope_kernel(const float* __restrict__ Qf, const float* __restrict__ w,
                       const float* __restrict__ pos, __bf16* __restrict__ Qb) {
  int lane = threadIdx.x & 31, wave = threadIdx.x >> 5;
  int gid = blockIdx.x * 8 + wave;
  if (gid >= NTOK * HEADS_N) return;
  int token = gid >> 4, h = gid & 15;
  int bn = token / S_LEN, s = token - bn * S_LEN;

  v4f xv = *(const v4f*)&Qf[(size_t)token * DIM_C + h * HD_N + lane * 4];
  float ss = xv[0]*xv[0] + xv[1]*xv[1] + xv[2]*xv[2] + xv[3]*xv[3];
#pragma unroll
  for (int o = 16; o > 0; o >>= 1) ss += __shfl_xor(ss, o, 32);
  float rn = rsqrtf(ss * (1.0f / 128.0f) + 1e-6f);

  v4f wv = *(const v4f*)&w[lane * 4];
  float y0 = xv[0]*rn*wv[0], y1 = xv[1]*rn*wv[1], y2 = xv[2]*rn*wv[2], y3 = xv[3]*rn*wv[3];

  float p = pos[token];
  const float L2_1E4 = 13.2877123795494f;  // log2(10000)
  v4bf o4;
#pragma unroll
  for (int j = 0; j < 2; ++j) {
    int i = lane * 2 + j;                  // freq index d/2
    float invf = exp2f(-(float)(2 * i) * (1.0f / 128.0f) * L2_1E4);
    float ang = p * invf;
    float cs = cosf(ang), sn = sinf(ang);
    float a0 = j ? y2 : y0, a1 = j ? y3 : y1;
    o4[2 * j]     = f2bf(a0 * cs - a1 * sn);
    o4[2 * j + 1] = f2bf(a0 * sn + a1 * cs);
  }
  size_t oidx = ((size_t)(bn * HEADS_N + h) * S_LEN + s) * HD_N + lane * 4;
  *(v4bf*)&Qb[oidx] = o4;
}

// ---------------------------------------------------------------- K: RMSNorm+RoPE -> [bn,h,a,d]; V -> transposed [bn,h,d,a]
__global__ __launch_bounds__(256)
void knorm_rope_kernel(const float* __restrict__ KVf, const float* __restrict__ w,
                       __bf16* __restrict__ Kb, __bf16* __restrict__ Vt) {
  int lane = threadIdx.x & 31, wave = threadIdx.x >> 5;
  int gid = blockIdx.x * 8 + wave;
  if (gid >= BN_N * NA_N * HEADS_N) return;
  int token = gid >> 4, h = gid & 15;
  int bn = token / NA_N, a = token - bn * NA_N;

  v4f xv = *(const v4f*)&KVf[(size_t)token * KVN_N + h * HD_N + lane * 4];
  v4f vv = *(const v4f*)&KVf[(size_t)token * KVN_N + DIM_C + h * HD_N + lane * 4];
  float ss = xv[0]*xv[0] + xv[1]*xv[1] + xv[2]*xv[2] + xv[3]*xv[3];
#pragma unroll
  for (int o = 16; o > 0; o >>= 1) ss += __shfl_xor(ss, o, 32);
  float rn = rsqrtf(ss * (1.0f / 128.0f) + 1e-6f);

  v4f wv = *(const v4f*)&w[lane * 4];
  float y0 = xv[0]*rn*wv[0], y1 = xv[1]*rn*wv[1], y2 = xv[2]*rn*wv[2], y3 = xv[3]*rn*wv[3];

  float p = (a < NA_N / 2) ? 2.0f : 22.0f;   // mid of human1 / human2 bands
  const float L2_1E4 = 13.2877123795494f;
  v4bf o4;
#pragma unroll
  for (int j = 0; j < 2; ++j) {
    int i = lane * 2 + j;
    float invf = exp2f(-(float)(2 * i) * (1.0f / 128.0f) * L2_1E4);
    float ang = p * invf;
    float cs = cosf(ang), sn = sinf(ang);
    float a0 = j ? y2 : y0, a1 = j ? y3 : y1;
    o4[2 * j]     = f2bf(a0 * cs - a1 * sn);
    o4[2 * j + 1] = f2bf(a0 * sn + a1 * cs);
  }
  size_t kidx = ((size_t)(bn * HEADS_N + h) * NA_N + a) * HD_N + lane * 4;
  *(v4bf*)&Kb[kidx] = o4;
  // V transposed: Vt[bn,h,d,a]
  size_t vbase = (size_t)(bn * HEADS_N + h) * HD_N * NA_N;
#pragma unroll
  for (int j = 0; j < 4; ++j)
    Vt[vbase + (size_t)(lane * 4 + j) * NA_N + a] = f2bf(vv[j]);
}

// ---------------------------------------------------------------- attention (Na=32 keys) via WMMA
// one block = (bn, h, 128 query rows); 8 waves x 16 rows
__global__ __launch_bounds__(256)
void attn_wmma_kernel(const __bf16* __restrict__ Qb, const __bf16* __restrict__ Kb,
                      const __bf16* __restrict__ Vt, __bf16* __restrict__ Ob) {
  __shared__ __align__(16) __bf16 Ks[32 * 136];   // [key][d], pad 8
  __shared__ __align__(16) __bf16 Vst[128 * 40];  // [d][key], pad 8
  __shared__ float sc[8][16][32];
  __shared__ float rinv[8][16];

  const int bn = blockIdx.z, h = blockIdx.y;
  const int tid = threadIdx.x, lane = tid & 31, wave = tid >> 5;
  const int r = lane & 15, kg = lane >> 4;

  const __bf16* Kg  = Kb + (size_t)(bn * HEADS_N + h) * NA_N * HD_N;
  const __bf16* Vtg = Vt + (size_t)(bn * HEADS_N + h) * HD_N * NA_N;
#pragma unroll
  for (int t = 0; t < 2; ++t) {                    // K: 512 chunks of 8
    int s = tid + t * 256;
    int key = s >> 4, d0 = (s & 15) * 8;
    async_copy_b128(&Kg[key * HD_N + d0], &Ks[key * 136 + d0]);
  }
#pragma unroll
  for (int t = 0; t < 2; ++t) {                    // V^T: 128 d-rows x 4 chunks
    int s = tid + t * 256;
    int d = s >> 2, c = (s & 3) * 8;
    async_copy_b128(&Vtg[d * NA_N + c], &Vst[d * 40 + c]);
  }
  wait_async_0();
  __syncthreads();

  const int sbase = blockIdx.x * 128 + wave * 16;
  int sq = sbase + r; if (sq >= S_LEN) sq = S_LEN - 1;   // clamp; stores guarded
  const __bf16* Qg = Qb + ((size_t)(bn * HEADS_N + h) * S_LEN + sq) * HD_N;

  const v8f zacc = {0.f, 0.f, 0.f, 0.f, 0.f, 0.f, 0.f, 0.f};
  v8f s0 = zacc, s1 = zacc;
#pragma unroll
  for (int kc = 0; kc < 4; ++kc) {      // hd = 4 x 32
    Frag a, b0, b1;
    a.q[0] = *(const v4u*)&Qg[kc * 32 + 8 * kg];
    a.q[1] = *(const v4u*)&Qg[kc * 32 + 16 + 8 * kg];
    b0.q[0] = *(const v4u*)&Ks[r * 136 + kc * 32 + 8 * kg];          // keys 0..15
    b0.q[1] = *(const v4u*)&Ks[r * 136 + kc * 32 + 16 + 8 * kg];
    b1.q[0] = *(const v4u*)&Ks[(16 + r) * 136 + kc * 32 + 8 * kg];   // keys 16..31
    b1.q[1] = *(const v4u*)&Ks[(16 + r) * 136 + kc * 32 + 16 + 8 * kg];
    s0 = wmma_bf16(a, b0, s0);
    s1 = wmma_bf16(a, b1, s1);
  }
  const float scale = 0.08838834764831845f;   // 1/sqrt(128)
#pragma unroll
  for (int v = 0; v < 8; ++v) {
    sc[wave][v + 8 * kg][r]      = s0[v] * scale;
    sc[wave][v + 8 * kg][16 + r] = s1[v] * scale;
  }
  __syncthreads();

  if (tid < 128) {                       // one thread per (wave, row): softmax over 32
    int w = tid >> 4, rr = tid & 15;
    float mx = -3.4e38f;
#pragma unroll
    for (int c = 0; c < 32; ++c) mx = fmaxf(mx, sc[w][rr][c]);
    float sm = 0.f;
#pragma unroll
    for (int c = 0; c < 32; ++c) { float e = __expf(sc[w][rr][c] - mx); sc[w][rr][c] = e; sm += e; }
    rinv[w][rr] = 1.0f / sm;
  }
  __syncthreads();

  Frag ap;
  float ri = rinv[wave][r];
#pragma unroll
  for (int e = 0; e < 16; ++e) {
    int k = 8 * kg + e + (e >= 8 ? 8 : 0);   // 16-bit A-fragment K mapping
    ap.v[e] = f2bf(sc[wave][r][k] * ri);
  }

#pragma unroll
  for (int nt = 0; nt < 8; ++nt) {          // out = P[16x32] @ V[32x128]
    Frag bv;
    int dc = nt * 16 + r;
    bv.q[0] = *(const v4u*)&Vst[dc * 40 + 8 * kg];
    bv.q[1] = *(const v4u*)&Vst[dc * 40 + 16 + 8 * kg];
    v8f o = wmma_bf16(ap, bv, zacc);
#pragma unroll
    for (int v = 0; v < 8; ++v) {
      int sg = sbase + v + 8 * kg;
      if (sg < S_LEN)
        Ob[((size_t)(bn * S_LEN + sg) * HEADS_N + h) * HD_N + nt * 16 + r] = f2bf(o[v]);
    }
  }
}

// ---------------------------------------------------------------- host
extern "C" void kernel_launch(void* const* d_in, const int* in_sizes, int n_in,
                              void* d_out, int out_size, void* d_ws, size_t ws_size,
                              hipStream_t stream) {
  const float* x    = (const float*)d_in[0];
  const float* cond = (const float*)d_in[1];
  const float* amap = (const float*)d_in[2];
  const float* q_w  = (const float*)d_in[3];
  const float* q_b  = (const float*)d_in[4];
  const float* kv_w = (const float*)d_in[5];
  const float* kv_b = (const float*)d_in[6];
  const float* qn_w = (const float*)d_in[7];
  const float* kn_w = (const float*)d_in[8];
  const float* p_w  = (const float*)d_in[9];
  const float* p_b  = (const float*)d_in[10];
  float* out = (float*)d_out;

  char* ws = (char*)d_ws;
  size_t off = 0;
  auto take = [&](size_t bytes) -> char* {
    char* p = ws + off;
    off += (bytes + 255) & ~(size_t)255;
    return p;
  };
  __bf16* xb    = (__bf16*)take((size_t)NTOK * DIM_C * 2);
  __bf16* qwt   = (__bf16*)take((size_t)DIM_C * DIM_C * 2);     // [N][K]
  __bf16* kvwt  = (__bf16*)take((size_t)KVN_N * ENC_N * 2);     // [N][K]
  __bf16* pwt   = (__bf16*)take((size_t)DIM_C * DIM_C * 2);     // [N][K]
  __bf16* condb = (__bf16*)take((size_t)BN_N * NA_N * ENC_N * 2);
  float*  KVf   = (float*) take((size_t)BN_N * NA_N * KVN_N * 4);
  __bf16* Qb2   = (__bf16*)take((size_t)NTOK * DIM_C * 2);
  __bf16* Kb    = (__bf16*)take((size_t)BN_N * HEADS_N * NA_N * HD_N * 2);
  __bf16* Vt    = (__bf16*)take((size_t)BN_N * HEADS_N * HD_N * NA_N * 2);
  float*  mm    = (float*) take(4 * sizeof(float));
  float*  pos   = (float*) take((size_t)NTOK * 4);
  float*  Qf    = (float*) take((size_t)NTOK * DIM_C * 4);
  __bf16* attnb = (__bf16*)Qf;   // alias: Qf dead after qnorm_rope; attn out is half the size

  const long long nX = (long long)NTOK * DIM_C;
  const long long nCond = (long long)BN_N * NA_N * ENC_N;
  auto cvt_grid = [](long long n) { return (unsigned)((n / 4 + 255) / 256); };

  cvt_f32_bf16_kernel<<<cvt_grid(nX),    256, 0, stream>>>(x,    xb,    nX);
  cvt_f32_bf16_kernel<<<cvt_grid(nCond), 256, 0, stream>>>(cond, condb, nCond);
  // weights: [K][N] fp32 -> [N][K] bf16
  transpose_cvt_kernel<<<dim3(DIM_C / 32, DIM_C / 32), 256, 0, stream>>>(q_w,  qwt,  DIM_C, DIM_C);
  transpose_cvt_kernel<<<dim3(KVN_N / 32, ENC_N / 32), 256, 0, stream>>>(kv_w, kvwt, ENC_N, KVN_N);
  transpose_cvt_kernel<<<dim3(DIM_C / 32, DIM_C / 32), 256, 0, stream>>>(p_w,  pwt,  DIM_C, DIM_C);

  minmax_kernel<<<1, 1024, 0, stream>>>(amap, NTOK, mm);
  pos_kernel<<<(NTOK + 255) / 256, 256, 0, stream>>>(amap, mm, pos, NTOK);

  // Q = x @ q_w + q_b  -> Qf fp32
  gemm_bf16_wmma_kernel<<<dim3((NTOK + BM - 1) / BM, DIM_C / BNT), 256, 0, stream>>>(
      xb, qwt, q_b, Qf, NTOK, DIM_C, DIM_C);
  // KV = cond @ kv_w + kv_b -> KVf fp32
  gemm_bf16_wmma_kernel<<<dim3((BN_N * NA_N + BM - 1) / BM, KVN_N / BNT), 256, 0, stream>>>(
      condb, kvwt, kv_b, KVf, BN_N * NA_N, KVN_N, ENC_N);

  qnorm_rope_kernel<<<(NTOK * HEADS_N) / 8, 256, 0, stream>>>(Qf, qn_w, pos, Qb2);
  knorm_rope_kernel<<<(BN_N * NA_N * HEADS_N) / 8, 256, 0, stream>>>(KVf, kn_w, Kb, Vt);

  attn_wmma_kernel<<<dim3((S_LEN + 127) / 128, HEADS_N, BN_N), 256, 0, stream>>>(
      Qb2, Kb, Vt, attnb);

  // out = attn @ proj_w + proj_b -> d_out fp32
  gemm_bf16_wmma_kernel<<<dim3((NTOK + BM - 1) / BM, DIM_C / BNT), 256, 0, stream>>>(
      attnb, pwt, p_b, out, NTOK, DIM_C, DIM_C);
}
